// LSTM_25048249270394
// MI455X (gfx1250) — matrix operand order repaired and checked by reading, not compiled, not run on hardware
//
#include <hip/hip_runtime.h>
#include <hip/hip_bf16.h>
#include <stdint.h>

// Problem constants (from reference)
#define PRED_LEN 24
#define F_  64
#define I_  128
#define H_  512
#define B_  1024
#define S_  168

typedef _Float16 half_t;
typedef __attribute__((ext_vector_type(16))) _Float16 v16h;
typedef __attribute__((ext_vector_type(8)))  _Float16 v8h;
typedef __attribute__((ext_vector_type(8)))  float    v8f;

#define WMMA_F16(A, B, C) \
  __builtin_amdgcn_wmma_f32_16x16x32_f16(false, (A), false, (B), (short)0, (C), false, false)

// ---------------------------------------------------------------------------
// Fragment loaders (wave32, v_wmma_f32_16x16x32_f16 layouts per CDNA5 ISA)
// ---------------------------------------------------------------------------
// A: 16x32 f16 tile from row-major [rows, ld]; lane l holds row (l&15);
// lanes 0-15 K=[k0,k0+8)+[k0+16,k0+24), lanes 16-31 K=+8 (ISA 7.12.2).
__device__ __forceinline__ v16h load_a_frag(const half_t* __restrict__ base,
                                            int ld, int k0, int lane) {
  const int r  = lane & 15;
  const int kh = (lane >> 4) << 3;      // 0 or 8
  const half_t* p = base + (size_t)r * ld + k0 + kh;
  v16h out;
  *reinterpret_cast<v8h*>(&out)       = *reinterpret_cast<const v8h*>(p);
  *(reinterpret_cast<v8h*>(&out) + 1) = *reinterpret_cast<const v8h*>(p + 16);
  return out;
}

// B: 32x16 f16 tile == W^T, W row-major [N, K]. Lane l holds column
// N = nbase+(l&15); lanes 0-15 K=[k0,k0+8)+[k0+16,k0+24), lanes 16-31 K=+8.
__device__ __forceinline__ v16h load_b_frag(const half_t* __restrict__ W,
                                            int ld, int nbase, int k0, int lane) {
  const int n  = nbase + (lane & 15);
  const int kk = k0 + ((lane >> 4) << 3); // +0 or +8
  const half_t* p = W + (size_t)n * ld + kk;
  v16h out;
  *reinterpret_cast<v8h*>(&out)       = *reinterpret_cast<const v8h*>(p);
  *(reinterpret_cast<v8h*>(&out) + 1) = *reinterpret_cast<const v8h*>(p + 16);
  return out;
}

__device__ __forceinline__ float sigmoidf_(float x) {
  return 1.0f / (1.0f + __expf(-x));
}

// ---------------------------------------------------------------------------
// Utility / precompute kernels (run once per launch)
// ---------------------------------------------------------------------------
__global__ void zero_u32_kernel(uint32_t* __restrict__ p, int n) {
  int i = blockIdx.x * blockDim.x + threadIdx.x;
  if (i < n) p[i] = 0u;
}

__global__ void cvt_f32_to_f16_kernel(const float* __restrict__ src,
                                      half_t* __restrict__ dst, int n) {
  int i = blockIdx.x * blockDim.x + threadIdx.x;
  if (i < n) dst[i] = (half_t)src[i];
}

// W0x[n][f] = sum_k Wih0[n][k] * Wenc[k][f]   (encoder folded into layer 0)
__global__ void fuse_enc_weights_kernel(const float* __restrict__ Wih0,
                                        const float* __restrict__ Wenc,
                                        half_t* __restrict__ W0x_h) {
  int idx = blockIdx.x * blockDim.x + threadIdx.x;   // 4H * F
  if (idx >= 4 * H_ * F_) return;
  const int n = idx / F_, f = idx % F_;
  float s = 0.0f;
  for (int k = 0; k < I_; ++k) s += Wih0[(size_t)n * I_ + k] * Wenc[(size_t)k * F_ + f];
  W0x_h[idx] = (half_t)s;
}

// bias0[n] = bih0[n] + bhh0[n] + sum_k Wih0[n][k] * b_enc[k]
__global__ void fuse_bias0_kernel(const float* __restrict__ Wih0,
                                  const float* __restrict__ benc,
                                  const float* __restrict__ bih0,
                                  const float* __restrict__ bhh0,
                                  float* __restrict__ bias0) {
  int n = blockIdx.x * blockDim.x + threadIdx.x;
  if (n >= 4 * H_) return;
  float s = bih0[n] + bhh0[n];
  for (int k = 0; k < I_; ++k) s += Wih0[(size_t)n * I_ + k] * benc[k];
  bias0[n] = s;
}

__global__ void add_bias_kernel(const float* __restrict__ a,
                                const float* __restrict__ b,
                                float* __restrict__ o, int n) {
  int i = blockIdx.x * blockDim.x + threadIdx.x;
  if (i < n) o[i] = a[i] + b[i];
}

// ---------------------------------------------------------------------------
// Fused LSTM cell: gates = xin@Wih^T + h@Whh^T + bias; update c, h.
// Each wave owns 32(M) x 16(N within H) and accumulates all 4 gates
// (N-tiles n, n+512, n+1024, n+1536): A-fragments shared across the 4 gate
// WMMAs and the cell nonlinearity is applied in-register (no gates buffer).
// grid (B/32, H/16), block 32 (one wave).
// ---------------------------------------------------------------------------
__global__ __launch_bounds__(32)
void lstm_cell_kernel(const half_t* __restrict__ xin, int ldx, int Kx, // [B,Kx]
                      const half_t* __restrict__ hr,            // [B,H] f16 (prev h)
                      const half_t* __restrict__ Wih,           // [4H,Kx] f16
                      const half_t* __restrict__ Whh,           // [4H,H] f16
                      const float*  __restrict__ bias,          // [4H] combined
                      float*  __restrict__ c,                   // [B,H] f32 in/out
                      half_t* __restrict__ hw)                  // [B,H] f16 out
{
  const int lane = threadIdx.x & 31;
  const int mt = blockIdx.x * 32;
  const int nt = blockIdx.y * 16;

  v8f acc[2][4] = {};

  // x contribution (K = Kx)
  for (int k0 = 0; k0 < Kx; k0 += 32) {
    v16h a0 = load_a_frag(xin + (size_t)mt * ldx, ldx, k0, lane);
    v16h a1 = load_a_frag(xin + (size_t)(mt + 16) * ldx, ldx, k0, lane);
    #pragma unroll
    for (int g = 0; g < 4; ++g) {
      v16h b = load_b_frag(Wih + (size_t)g * H_ * Kx, Kx, nt, k0, lane);
      acc[0][g] = WMMA_F16(a0, b, acc[0][g]);
      acc[1][g] = WMMA_F16(a1, b, acc[1][g]);
    }
  }
  // h contribution (K = H)
  for (int k0 = 0; k0 < H_; k0 += 32) {
    v16h a0 = load_a_frag(hr + (size_t)mt * H_, H_, k0, lane);
    v16h a1 = load_a_frag(hr + (size_t)(mt + 16) * H_, H_, k0, lane);
    #pragma unroll
    for (int g = 0; g < 4; ++g) {
      v16h b = load_b_frag(Whh + (size_t)g * H_ * H_, H_, nt, k0, lane);
      acc[0][g] = WMMA_F16(a0, b, acc[0][g]);
      acc[1][g] = WMMA_F16(a1, b, acc[1][g]);
    }
  }

  const int n  = nt + (lane & 15);
  const int mo = (lane >> 4) << 3;
  float bs[4];
  #pragma unroll
  for (int g = 0; g < 4; ++g) bs[g] = bias[g * H_ + n];

  #pragma unroll
  for (int sub = 0; sub < 2; ++sub) {
    #pragma unroll
    for (int r = 0; r < 8; ++r) {
      const int m = mt + sub * 16 + mo + r;
      const size_t idx = (size_t)m * H_ + n;
      const float iv = sigmoidf_(acc[sub][0][r] + bs[0]);
      const float fv = sigmoidf_(acc[sub][1][r] + bs[1]);
      const float gv = tanhf(acc[sub][2][r] + bs[2]);
      const float ov = sigmoidf_(acc[sub][3][r] + bs[3]);
      const float cv = fv * c[idx] + iv * gv;
      c[idx]  = cv;
      hw[idx] = (half_t)(ov * tanhf(cv));
    }
  }
}

// ---------------------------------------------------------------------------
// Decoder: out[B,F] = h1[B,H] @ W_dec^T + b_dec. Writes f32 strided slice of
// d_out ([B, PRED_LEN, F]) and the f16 AR feedback buffer.
// grid (B/32, F/16), block 32.
// ---------------------------------------------------------------------------
__global__ __launch_bounds__(32)
void dec_kernel(const half_t* __restrict__ h1,     // [B,H] f16
                const half_t* __restrict__ Wd,     // [F_,H_] f16
                const float*  __restrict__ bd,     // [F_]
                float* __restrict__ outp, int out_ld,
                half_t* __restrict__ xfb)          // [B,F_] f16
{
  const int lane = threadIdx.x & 31;
  const int mt = blockIdx.x * 32;
  const int nt = blockIdx.y * 16;
  v8f acc0 = {}, acc1 = {};
  for (int k0 = 0; k0 < H_; k0 += 32) {
    v16h a0 = load_a_frag(h1 + (size_t)mt * H_, H_, k0, lane);
    v16h a1 = load_a_frag(h1 + (size_t)(mt + 16) * H_, H_, k0, lane);
    v16h b  = load_b_frag(Wd, H_, nt, k0, lane);
    acc0 = WMMA_F16(a0, b, acc0);
    acc1 = WMMA_F16(a1, b, acc1);
  }
  const int n  = nt + (lane & 15);
  const int mo = (lane >> 4) << 3;
  const float bn = bd[n];
  #pragma unroll
  for (int r = 0; r < 8; ++r) {
    const int m0 = mt + mo + r;
    const int m1 = mt + 16 + mo + r;
    const float v0 = acc0[r] + bn;
    const float v1 = acc1[r] + bn;
    outp[(size_t)m0 * out_ld + n] = v0;
    outp[(size_t)m1 * out_ld + n] = v1;
    xfb[(size_t)m0 * F_ + n] = (half_t)v0;
    xfb[(size_t)m1 * F_ + n] = (half_t)v1;
  }
}

// ---------------------------------------------------------------------------
// Launch orchestration
// ---------------------------------------------------------------------------
extern "C" void kernel_launch(void* const* d_in, const int* in_sizes, int n_in,
                              void* d_out, int out_size, void* d_ws, size_t ws_size,
                              hipStream_t stream) {
  (void)in_sizes; (void)n_in; (void)out_size; (void)ws_size;

  const float* x     = (const float*)d_in[0];
  const float* W_enc = (const float*)d_in[1];
  const float* b_enc = (const float*)d_in[2];
  const float* Wih0  = (const float*)d_in[3];
  const float* Whh0  = (const float*)d_in[4];
  const float* bih0  = (const float*)d_in[5];
  const float* bhh0  = (const float*)d_in[6];
  const float* Wih1  = (const float*)d_in[7];
  const float* Whh1  = (const float*)d_in[8];
  const float* bih1  = (const float*)d_in[9];
  const float* bhh1  = (const float*)d_in[10];
  const float* W_dec = (const float*)d_in[11];
  const float* b_dec = (const float*)d_in[12];
  float* out = (float*)d_out;

  // Workspace carve-up (~36 MB; all chunks 256B aligned).
  char* ws = (char*)d_ws;
  size_t off = 0;
  auto carve = [&](size_t bytes) -> char* {
    char* p = ws + off;
    off += (bytes + 255) & ~(size_t)255;
    return p;
  };
  // State region first (zeroed each call): h ping-pongs (f16) + c (f32).
  half_t* h0buf[2] = { (half_t*)carve((size_t)B_ * H_ * 2),
                       (half_t*)carve((size_t)B_ * H_ * 2) };
  half_t* h1buf[2] = { (half_t*)carve((size_t)B_ * H_ * 2),
                       (half_t*)carve((size_t)B_ * H_ * 2) };
  float* c0 = (float*)carve((size_t)B_ * H_ * 4);
  float* c1 = (float*)carve((size_t)B_ * H_ * 4);
  const size_t state_bytes = off;           // contiguous -> one zero kernel

  half_t* xfb    = (half_t*)carve((size_t)B_ * F_ * 2);
  half_t* W0x_h  = (half_t*)carve((size_t)4 * H_ * F_ * 2);   // fused Wih0*W_enc
  half_t* Whh0_h = (half_t*)carve((size_t)4 * H_ * H_ * 2);
  half_t* Wih1_h = (half_t*)carve((size_t)4 * H_ * H_ * 2);
  half_t* Whh1_h = (half_t*)carve((size_t)4 * H_ * H_ * 2);
  half_t* Wdec_h = (half_t*)carve((size_t)F_ * H_ * 2);
  float*  bias0  = (float*)carve((size_t)4 * H_ * 4);         // bih0+bhh0+Wih0*b_enc
  float*  bias1  = (float*)carve((size_t)4 * H_ * 4);         // bih1+bhh1
  half_t* x_h    = (half_t*)carve((size_t)B_ * S_ * F_ * 2);

  // 1) zero LSTM state (deterministic: state rebuilt every call)
  {
    int nw = (int)(state_bytes / 4);
    zero_u32_kernel<<<(nw + 255) / 256, 256, 0, stream>>>((uint32_t*)ws, nw);
  }

  // 2) precompute fused weights / biases + f16 conversions
  auto cvt = [&](const float* s, half_t* d, int n) {
    cvt_f32_to_f16_kernel<<<(n + 255) / 256, 256, 0, stream>>>(s, d, n);
  };
  fuse_enc_weights_kernel<<<(4 * H_ * F_ + 255) / 256, 256, 0, stream>>>(
      Wih0, W_enc, W0x_h);
  fuse_bias0_kernel<<<(4 * H_ + 255) / 256, 256, 0, stream>>>(
      Wih0, b_enc, bih0, bhh0, bias0);
  add_bias_kernel<<<(4 * H_ + 255) / 256, 256, 0, stream>>>(bih1, bhh1, bias1, 4 * H_);
  cvt(Whh0,  Whh0_h, 4 * H_ * H_);
  cvt(Wih1,  Wih1_h, 4 * H_ * H_);
  cvt(Whh1,  Whh1_h, 4 * H_ * H_);
  cvt(W_dec, Wdec_h, F_ * H_);
  cvt(x,     x_h,    B_ * S_ * F_);

  // 3) recurrent rollout: 168 teacher-forced + 23 autoregressive steps.
  //    Encoder eliminated algebraically -> only 2 dependent kernels per step.
  const dim3 blk(32);
  const dim3 gCell(B_ / 32, H_ / 16);
  const dim3 gDec(B_ / 32, F_ / 16);

  int p0 = 0, p1 = 0;
  for (int t = 0; t < S_ + PRED_LEN - 1; ++t) {
    const half_t* xin;
    int ldx;
    if (t < S_) { xin = x_h + (size_t)t * F_; ldx = S_ * F_; }  // x[:, t, :]
    else        { xin = xfb;                  ldx = F_; }        // AR feedback

    // layer 0: gates = x_t @ (Wih0*W_enc)^T + h0 @ Whh0^T + bias0
    lstm_cell_kernel<<<gCell, blk, 0, stream>>>(
        xin, ldx, F_, h0buf[p0], W0x_h, Whh0_h, bias0, c0, h0buf[1 - p0]);

    // layer 1: gates = h0 @ Wih1^T + h1 @ Whh1^T + bias1
    lstm_cell_kernel<<<gCell, blk, 0, stream>>>(
        h0buf[1 - p0], H_, H_, h1buf[p1], Wih1_h, Whh1_h, bias1, c1, h1buf[1 - p1]);

    p0 ^= 1;
    p1 ^= 1;

    const int j = t - (S_ - 1);   // output slice index
    if (j >= 0) {
      dec_kernel<<<gDec, blk, 0, stream>>>(
          h1buf[p1], Wdec_h, b_dec, out + (size_t)j * F_, PRED_LEN * F_, xfb);
    }
  }
}
